// Attention_42253888258536
// MI455X (gfx1250) — compile-verified, matrix operand order checked by reading, not run on hardware
//
#include <hip/hip_runtime.h>
#include <hip/hip_bf16.h>

typedef __attribute__((ext_vector_type(16))) _Float16 v16h;
typedef __attribute__((ext_vector_type(8)))  float    v8f;
typedef int v4i __attribute__((vector_size(16)));

#ifndef __has_builtin
#define __has_builtin(x) 0
#endif

#if __has_builtin(__builtin_amdgcn_global_load_async_to_lds_b128) && \
    __has_builtin(__builtin_amdgcn_s_wait_asynccnt)
#define HAS_ASYNC_LDS 1
#else
#define HAS_ASYNC_LDS 0
#endif

#define DEVI static __device__ __forceinline__

DEVI v8f wmma_f16(v16h a, v16h b, v8f c) {
  // D = A(16x32 f16) * B(32x16 f16) + C(16x16 f32)
  return __builtin_amdgcn_wmma_f32_16x16x32_f16(
      /*neg_a=*/false, a, /*neg_b=*/false, b,
      /*c_mod=*/(short)0, c, /*reuse_a=*/false, /*reuse_b=*/false);
}

// 16-byte global->LDS copy. Async (ASYNCcnt DMA, global_load_async_to_lds_b128)
// when the builtin exists, otherwise a synchronous VGPR round trip.
// Address-space pointers are built via integer casts: LDS generic pointers
// carry the LDS offset in the low 32 bits; global generic pointers are the VA.
DEVI void copy_b128_to_lds(const void* g, void* l) {
#if HAS_ASYNC_LDS
  typedef __attribute__((address_space(1))) v4i gv4i;
  typedef __attribute__((address_space(3))) v4i lv4i;
  __builtin_amdgcn_global_load_async_to_lds_b128(
      (gv4i*)(unsigned long long)g,
      (lv4i*)(unsigned int)(unsigned long long)l,
      /*offset=*/0, /*cpol=*/0);
#else
  *(v4i*)l = *(const v4i*)g;
#endif
}

DEVI void copy_to_lds_wait() {
#if HAS_ASYNC_LDS
  __builtin_amdgcn_s_wait_asynccnt(0);
#endif
}

// ---------------------------------------------------------------------------
// Kernel 1: qkv = x @ qkv_w^T + qkv_b, fused RoPE on q,k.
//   Q,K scattered to [B*H, L, hd] f16;  V scattered TRANSPOSED to
//   [B*H, hd, L] f16 so the attention kernel can async-DMA contiguous
//   B-fragments for P*V. One wave computes a 16x16 output tile.
// ---------------------------------------------------------------------------
__global__ __launch_bounds__(256)
void qkv_rope_kernel(const float* __restrict__ x,      // [B*L, 512]
                     const float* __restrict__ w,      // [1536, 512]
                     const float* __restrict__ bias,   // [1536]
                     _Float16* __restrict__ Qws,
                     _Float16* __restrict__ Kws,
                     _Float16* __restrict__ Vtws)
{
  const int lane  = threadIdx.x & 31;
  const int wave  = threadIdx.x >> 5;
  const int mtile = blockIdx.x;               // 0..511  (rows b*L+l, /16)
  const int ntile = blockIdx.y * 8 + wave;    // 0..95   (out channel /16)
  const int ln    = lane & 15;
  const int klo   = (lane < 16) ? 0 : 8;
  const int C     = 512;

  const int arow = mtile * 16 + ln;
  const int brow = ntile * 16 + ln;

  v8f acc = {};
  for (int k0 = 0; k0 < C; k0 += 32) {
    v16h a, bf;
    const float* ap = x + (size_t)arow * C + k0 + klo;
    const float* bp = w + (size_t)brow * C + k0 + klo;
#pragma unroll
    for (int i = 0; i < 8; ++i) {
      a[i]      = (_Float16)ap[i];
      a[8 + i]  = (_Float16)ap[16 + i];
      bf[i]     = (_Float16)bp[i];
      bf[8 + i] = (_Float16)bp[16 + i];
    }
    acc = wmma_f16(a, bf, acc);
  }

  // Epilogue: bias, RoPE (q,k only), scatter.
  const int c     = ntile * 16 + ln;     // global out channel 0..1535
  const float bv  = bias[c];
  const int which = c >> 9;              // 0=q, 1=k, 2=v (tile never straddles)
  const int h     = (c & 511) >> 6;
  const int d     = c & 63;
  const int j     = d & 31;
  // 10000^(-2j/64) = exp2(-j * (2/64) * log2(10000)); hardware exp is base-2
  const float inv_freq = exp2f((float)j * -0.41524100776293f);
  const float sgn = (d & 1) ? 1.0f : -1.0f;
  const int rbase = mtile * 16 + ((lane < 16) ? 0 : 8);

#pragma unroll
  for (int r = 0; r < 8; ++r) {
    float v = acc[r] + bv;
    const int row = rbase + r;        // = b*4096 + l
    const int l   = row & 4095;
    const int b   = row >> 12;
    const int bh  = b * 8 + h;
    // interleaved rotate-half: partner channel d^1 lives in lane^1
    const float partner = __shfl_xor(v, 1, 32);
    if (which < 2) {
      const float ang = (float)l * inv_freq;
      v = v * __cosf(ang) + sgn * partner * __sinf(ang);
    }
    if (which == 0)
      Qws[(size_t)bh * 4096 * 64 + (size_t)l * 64 + d] = (_Float16)v;
    else if (which == 1)
      Kws[(size_t)bh * 4096 * 64 + (size_t)l * 64 + d] = (_Float16)v;
    else  // V stored transposed: [bh][d][l] (8 consecutive l per lane)
      Vtws[(size_t)bh * 4096 * 64 + (size_t)d * 4096 + l] = (_Float16)v;
  }
}

// ---------------------------------------------------------------------------
// Kernel 2: flash attention. Block = 8 waves sharing one (b,h); each wave owns
// a 16-query tile. K ([key][d]) and V^T ([d][key]) 32-key chunks are staged
// into LDS with async b128 DMA (ASYNCcnt) and consumed as WMMA fragments.
// Online softmax keeps per-half-wave row stats; P is converted from C-layout
// to A-fragment via a per-wave LDS bounce guarded by s_wait_dscnt.
// ---------------------------------------------------------------------------
__global__ __launch_bounds__(256)
void flash_attn_kernel(const _Float16* __restrict__ Qws,   // [bh][l][d]
                       const _Float16* __restrict__ Kws,   // [bh][l][d]
                       const _Float16* __restrict__ Vtws,  // [bh][d][l]
                       _Float16* __restrict__ Ows)         // [bh][l][d]
{
  __shared__ __align__(16) _Float16 Klds[32 * 64];      // [key][d]
  __shared__ __align__(16) _Float16 Vlds[64 * 32];      // [d][key]
  __shared__ __align__(16) _Float16 Plds[8][16 * 32];   // per-wave [row][key]

  const int tid    = threadIdx.x;
  const int lane   = tid & 31;
  const int wave   = tid >> 5;
  const int bh     = blockIdx.x >> 5;           // 0..15
  const int qblk   = blockIdx.x & 31;           // 0..31
  const int q_tile = qblk * 8 + wave;           // 0..255
  const int ln     = lane & 15;
  const int klo    = (lane < 16) ? 0 : 8;
  const size_t base = (size_t)bh * 4096 * 64;

  // Per-thread staging assignments (16B granules, 256 of each per chunk)
  const int kkey = tid >> 3, kseg = tid & 7;    // K:  [key][seg*8 .. +8)
  const int vd   = tid >> 2, vseg = tid & 3;    // Vt: [d][seg*8 .. +8)
  const _Float16* Kg = Kws  + base + (size_t)kkey * 64 + kseg * 8;
  const _Float16* Vg = Vtws + base + (size_t)vd * 4096 + vseg * 8;
  _Float16* Kl = &Klds[kkey * 64 + kseg * 8];
  _Float16* Vl = &Vlds[vd * 32 + vseg * 8];

  // Q fragments for hd-chunks k0 = 0, 32
  v16h qf[2];
  {
    const _Float16* qp = Qws + base + (size_t)(q_tile * 16 + ln) * 64;
#pragma unroll
    for (int t = 0; t < 2; ++t) {
      const int kb = t * 32 + klo;
#pragma unroll
      for (int i = 0; i < 8; ++i) {
        qf[t][i]     = qp[kb + i];
        qf[t][8 + i] = qp[kb + 16 + i];
      }
    }
  }

  v8f o0 = {}, o1 = {}, o2 = {}, o3 = {};
  float m_[8], l_[8];
#pragma unroll
  for (int r = 0; r < 8; ++r) { m_[r] = -1e30f; l_[r] = 0.0f; }

  const float scale = 0.125f;  // 64^-0.5

  for (int kc = 0; kc < 4096; kc += 32) {
    __syncthreads();  // previous chunk's LDS reads done before overwrite
    copy_b128_to_lds(Kg + (size_t)kc * 64, Kl);   // K rows advance by kc keys
    copy_b128_to_lds(Vg + kc, Vl);                // Vt rows advance by kc cols
    copy_to_lds_wait();
    __syncthreads();

    if (kc + 32 < 4096) {                         // prefetch next chunk
      __builtin_prefetch(Kg + (size_t)(kc + 32) * 64, 0, 1);
      __builtin_prefetch(Vg + kc + 32, 0, 1);
    }

    // S = Q * K^T : two 16-key subtiles, contraction over hd in 2 chunks
    v8f s0 = {}, s1 = {};
#pragma unroll
    for (int t = 0; t < 2; ++t) {
      v16h kb0, kb1;
      const _Float16* kp0 = &Klds[(0  + ln) * 64 + t * 32 + klo];
      const _Float16* kp1 = &Klds[(16 + ln) * 64 + t * 32 + klo];
#pragma unroll
      for (int i = 0; i < 8; ++i) {
        kb0[i] = kp0[i];  kb0[8 + i] = kp0[16 + i];
        kb1[i] = kp1[i];  kb1[8 + i] = kp1[16 + i];
      }
      s0 = wmma_f16(qf[t], kb0, s0);
      s1 = wmma_f16(qf[t], kb1, s1);
    }

    // Online softmax per row (rows live in 16-lane halves)
    _Float16* Pw = Plds[wave];
#pragma unroll
    for (int r = 0; r < 8; ++r) {
      const float a0 = s0[r] * scale;
      const float a1 = s1[r] * scale;
      float mx = fmaxf(a0, a1);
#pragma unroll
      for (int off = 8; off >= 1; off >>= 1)
        mx = fmaxf(mx, __shfl_xor(mx, off, 32));
      const float mn    = fmaxf(m_[r], mx);
      const float alpha = __expf(m_[r] - mn);
      const float p0    = __expf(a0 - mn);
      const float p1    = __expf(a1 - mn);
      float rs = p0 + p1;
#pragma unroll
      for (int off = 8; off >= 1; off >>= 1)
        rs += __shfl_xor(rs, off, 32);
      l_[r] = l_[r] * alpha + rs;
      m_[r] = mn;
      o0[r] *= alpha; o1[r] *= alpha; o2[r] *= alpha; o3[r] *= alpha;
      const int prow = r + ((lane < 16) ? 0 : 8);
      Pw[prow * 32 + ln]      = (_Float16)p0;
      Pw[prow * 32 + 16 + ln] = (_Float16)p1;
    }
    asm volatile("s_wait_dscnt 0" ::: "memory");  // P bounce visible in-wave

    // P A-fragment (16 queries x 32 keys)
    v16h pf;
    {
      const _Float16* pp = &Pw[ln * 32 + klo];
#pragma unroll
      for (int i = 0; i < 8; ++i) { pf[i] = pp[i]; pf[8 + i] = pp[16 + i]; }
    }

    // O += P * V  (four hd tiles of 16 columns)
#pragma unroll
    for (int t = 0; t < 4; ++t) {
      v16h vb;
      const _Float16* vp = &Vlds[(t * 16 + ln) * 32 + klo];
#pragma unroll
      for (int i = 0; i < 8; ++i) { vb[i] = vp[i]; vb[8 + i] = vp[16 + i]; }
      if (t == 0)      o0 = wmma_f16(pf, vb, o0);
      else if (t == 1) o1 = wmma_f16(pf, vb, o1);
      else if (t == 2) o2 = wmma_f16(pf, vb, o2);
      else             o3 = wmma_f16(pf, vb, o3);
    }
  }

  // Normalize and store [bh][l][d] f16
#pragma unroll
  for (int r = 0; r < 8; ++r) {
    const float inv = 1.0f / l_[r];
    const int row = q_tile * 16 + r + ((lane < 16) ? 0 : 8);
    _Float16* op = Ows + base + (size_t)row * 64 + ln;
    op[0]  = (_Float16)(o0[r] * inv);
    op[16] = (_Float16)(o1[r] * inv);
    op[32] = (_Float16)(o2[r] * inv);
    op[48] = (_Float16)(o3[r] * inv);
  }
}

// ---------------------------------------------------------------------------
// Kernel 3: out = attn_out @ proj_w^T + proj_b. A rows gathered from
// [B,H,L,hd]; 8-element K segments are head-aligned so loads stay contiguous.
// ---------------------------------------------------------------------------
__global__ __launch_bounds__(256)
void proj_kernel(const _Float16* __restrict__ Aws,   // [B,H,L,64]
                 const float* __restrict__ w,        // [512,512]
                 const float* __restrict__ bias,     // [512]
                 float* __restrict__ out)            // [B*L,512]
{
  const int lane  = threadIdx.x & 31;
  const int wave  = threadIdx.x >> 5;
  const int mtile = blockIdx.x;               // 0..511
  const int ntile = blockIdx.y * 8 + wave;    // 0..31
  const int ln    = lane & 15;
  const int klo   = (lane < 16) ? 0 : 8;

  const int arow = mtile * 16 + ln;           // b*4096 + l
  const int b    = arow >> 12;
  const int l    = arow & 4095;
  const int brow = ntile * 16 + ln;

  v8f acc = {};
  for (int k0 = 0; k0 < 512; k0 += 32) {
    v16h a, bf;
#pragma unroll
    for (int half = 0; half < 2; ++half) {
      const int kb = k0 + klo + half * 16;    // 8-aligned, never crosses head
      const int h  = kb >> 6;
      const int d  = kb & 63;
      const _Float16* ap = Aws + ((size_t)((b * 8 + h) * 4096 + l)) * 64 + d;
#pragma unroll
      for (int i = 0; i < 8; ++i) a[half * 8 + i] = ap[i];
    }
    const float* bp = w + (size_t)brow * 512 + k0 + klo;
#pragma unroll
    for (int i = 0; i < 8; ++i) {
      bf[i]     = (_Float16)bp[i];
      bf[8 + i] = (_Float16)bp[16 + i];
    }
    acc = wmma_f16(a, bf, acc);
  }

  const int col   = ntile * 16 + ln;
  const float bv  = bias[col];
  const int rbase = mtile * 16 + ((lane < 16) ? 0 : 8);
#pragma unroll
  for (int r = 0; r < 8; ++r)
    out[(size_t)(rbase + r) * 512 + col] = acc[r] + bv;
}

// ---------------------------------------------------------------------------
extern "C" void kernel_launch(void* const* d_in, const int* in_sizes, int n_in,
                              void* d_out, int out_size, void* d_ws, size_t ws_size,
                              hipStream_t stream) {
  const float* x      = (const float*)d_in[0];  // [2,4096,512]
  const float* qkv_w  = (const float*)d_in[1];  // [1536,512]
  const float* qkv_b  = (const float*)d_in[2];  // [1536]
  const float* proj_w = (const float*)d_in[3];  // [512,512]
  const float* proj_b = (const float*)d_in[4];  // [512]
  float* out          = (float*)d_out;          // [2,4096,512]

  const size_t QKV_ELEMS = (size_t)2 * 8 * 4096 * 64;  // 4,194,304 per tensor
  _Float16* Qws  = (_Float16*)d_ws;
  _Float16* Kws  = Qws + QKV_ELEMS;
  _Float16* Vtws = Kws + QKV_ELEMS;
  _Float16* Ows  = Vtws + QKV_ELEMS;   // total 32 MB of workspace

  qkv_rope_kernel<<<dim3(512, 12), 256, 0, stream>>>(x, qkv_w, qkv_b,
                                                     Qws, Kws, Vtws);
  flash_attn_kernel<<<dim3(512), 256, 0, stream>>>(Qws, Kws, Vtws, Ows);
  proj_kernel<<<dim3(512, 4), 256, 0, stream>>>(Ows, proj_w, proj_b, out);
}
